// DynamicRNN_42941083026042
// MI455X (gfx1250) — matrix operand order repaired and checked by reading, not compiled, not run on hardware
//
#include <hip/hip_runtime.h>

// ---------------------------------------------------------------------------
// Dynamic LSTM for MI455X (gfx1250), wave32 + WMMA bf16.
//   B=128, T=1024, D=512, H=512.
// Persistent-kernel design: one launch loops over all 1024 timesteps with a
// grid-wide atomic barrier. Recurrent+input GEMM fused into one K=1024 bf16
// WMMA GEMM per step; weights converted once to bf16 (4 MB, L2-resident).
// K-loop unrolled x4 (proven no-spill steady state); B weights addressed as
// uniform base + 32-bit lane offset so the backend uses SGPR-base+VGPR-offset
// addressing with immediate k0 folds.
// ---------------------------------------------------------------------------

typedef __attribute__((ext_vector_type(8)))  float   v8f;
typedef __attribute__((ext_vector_type(8)))  __bf16  v8bf;
typedef __attribute__((ext_vector_type(16))) __bf16  v16bf;

union BfFrag { v16bf v; v8bf h[2]; };

#define Bc    128
#define Tc    1024
#define Dc    512
#define Hc    512
#define Kc    1024          // D + H
#define G4    2048          // 4*H
#define NBLK  32
#define PSTR  1032          // padded LDS row stride (bf16 elems): 2064 B, conflict-free

// workspace layout (bytes)
#define OFF_WC   0u                         // 2048*1024 bf16  = 4,194,304
#define OFF_BIAS 4194304u                   // 2048 f32        = 8,192
#define OFF_HBF  4202496u                   // 128*512 bf16    = 131,072
#define OFF_HF   4333568u                   // 128*512 f32     = 262,144
#define OFF_CF   4595712u                   // 128*512 f32     = 262,144
#define OFF_CNT  4857856u                   // 1 u32

__device__ __forceinline__ unsigned short f2bf(float f) {
  unsigned u = __builtin_bit_cast(unsigned, f);
  u += 0x7FFFu + ((u >> 16) & 1u);          // round-to-nearest-even
  return (unsigned short)(u >> 16);
}
__device__ __forceinline__ float sigmf(float x)    { return 1.0f / (1.0f + __expf(-x)); }
__device__ __forceinline__ float tanhfast(float x) { float e = __expf(2.0f * x); return 1.0f - 2.0f / (e + 1.0f); }

// ---------------------------------------------------------------------------
// Prep: bf16-convert [W_ih | W_hh] into K-major combined weights, sum biases,
// zero h/c state, reset barrier counter.
// ---------------------------------------------------------------------------
__global__ void lstm_prep(const float* __restrict__ Wih, const float* __restrict__ Whh,
                          const float* __restrict__ bih, const float* __restrict__ bhh,
                          unsigned short* __restrict__ Wc, float* __restrict__ bias,
                          unsigned short* __restrict__ hbf, float* __restrict__ hf,
                          float* __restrict__ cf, unsigned* __restrict__ cnt) {
  int idx = blockIdx.x * blockDim.x + threadIdx.x;
  if (idx < G4 * Kc) {
    int row = idx >> 10;                    // gate-output row 0..2047
    int k   = idx & (Kc - 1);               // combined K 0..1023
    float v = (k < Dc) ? Wih[row * Dc + k] : Whh[row * Hc + (k - Dc)];
    Wc[idx] = f2bf(v);
  }
  if (idx < G4)       bias[idx] = bih[idx] + bhh[idx];
  if (idx < Bc * Hc)  { hbf[idx] = 0; hf[idx] = 0.0f; cf[idx] = 0.0f; }
  if (idx == 0)       *cnt = 0u;
}

// ---------------------------------------------------------------------------
// Persistent LSTM kernel. 32 blocks x 256 threads (8 waves).
// task = blk*8 + wave; m = task>>5 (batch tile, shared by whole block),
// n = task&31 (column group). Wave computes i/f/g/o tiles at (m,n),
// sharing one A fragment per K-step across 4 WMMAs.
// ---------------------------------------------------------------------------
__global__ __launch_bounds__(256)
void lstm_persist(const float* __restrict__ X, const long long* __restrict__ lens,
                  const unsigned short* __restrict__ Wc, const float* __restrict__ bias,
                  unsigned short* __restrict__ hbf, float* __restrict__ hf,
                  float* __restrict__ cf, unsigned* __restrict__ cnt,
                  float* __restrict__ out, float* __restrict__ hfin, float* __restrict__ cfin) {
  __shared__ __align__(16) unsigned short panel[16 * PSTR];   // 16 rows x K=1024 bf16 (padded)

  const int tid  = threadIdx.x;
  const int blk  = blockIdx.x;
  const int wave = tid >> 5;
  const int lane = tid & 31;
  const int task = blk * 8 + wave;
  const int mt   = task >> 5;               // 0..7  batch-row tile
  const int nt   = task & 31;               // 0..31 column-group tile
  const int b0   = mt * 16;                 // first batch row of tile
  const int col  = nt * 16 + (lane & 15);   // this lane's hidden column j
  const int rbase = (lane < 16) ? 0 : 8;    // C-layout row offset for this lane half
  const int arow  = lane & 15;
  const int koffA = (lane < 16) ? 0 : 16;   // A-frag K split (matches B)
  const int koffB = (lane < 16) ? 0 : 16;   // B-frag K split

  // per-lane invariants: bias and 32-bit weight offsets (uniform base Wc)
  float bsum[4];
  unsigned woff[4];
  #pragma unroll
  for (int g = 0; g < 4; ++g) {
    bsum[g] = bias[g * Hc + col];
    woff[g] = (unsigned)(g * Hc + col) * (unsigned)Kc + (unsigned)koffB;
  }
  int lenr[8];
  #pragma unroll
  for (int r = 0; r < 8; ++r) lenr[r] = (int)lens[b0 + rbase + r];

  for (int t = 0; t < Tc; ++t) {
    // -------- stage A panel: rows b0..b0+15, K = [x_t (f32->bf16) | h (bf16)] ----
    for (int g4 = tid; g4 < 16 * 256; g4 += 256) {
      int row = g4 >> 8;
      int kq  = (g4 & 255) << 2;            // 4 elements per granule
      int b   = b0 + row;
      if (kq < Dc) {
        const float4 xv = *(const float4*)(X + (size_t)b * Tc * Dc + (size_t)t * Dc + kq);
        unsigned lo = (unsigned)f2bf(xv.x) | ((unsigned)f2bf(xv.y) << 16);
        unsigned hi = (unsigned)f2bf(xv.z) | ((unsigned)f2bf(xv.w) << 16);
        uint2 pk; pk.x = lo; pk.y = hi;
        *(uint2*)(void*)(panel + row * PSTR + kq) = pk;
      } else {
        *(uint2*)(void*)(panel + row * PSTR + kq) =
            *(const uint2*)(const void*)(hbf + b * Hc + (kq - Dc));
      }
    }
    __syncthreads();

    // -------- GEMM: gates(16x16 x4) += A(16xK) * W^T --------------------------
    v8f acc[4];
    #pragma unroll
    for (int g = 0; g < 4; ++g) acc[g] = v8f{};

    const unsigned short* abase = panel + arow * PSTR + koffA;
    #pragma unroll 4
    for (int k0 = 0; k0 < Kc; k0 += 32) {
      BfFrag A;
      A.h[0] = *(const v8bf*)(const void*)(abase + k0);
      A.h[1] = *(const v8bf*)(const void*)(abase + k0 + 8);
      #pragma unroll
      for (int g = 0; g < 4; ++g) {
        BfFrag Bf;
        const v8bf* wp = (const v8bf*)(const void*)(Wc + woff[g] + (unsigned)k0);
        Bf.h[0] = wp[0];
        Bf.h[1] = wp[1];
        acc[g] = __builtin_amdgcn_wmma_f32_16x16x32_bf16(
            false, A.v, false, Bf.v, (short)0, acc[g], false, false);
      }
    }

    // -------- LSTM cell, masked update, streamed output -----------------------
    #pragma unroll
    for (int r = 0; r < 8; ++r) {
      int b = b0 + rbase + r;
      float iv = sigmf   (acc[0][r] + bsum[0]);
      float fv = sigmf   (acc[1][r] + bsum[1]);
      float gv = tanhfast(acc[2][r] + bsum[2]);
      float ov = sigmf   (acc[3][r] + bsum[3]);
      float cprev = cf[(size_t)b * Hc + col];
      float cn = fv * cprev + iv * gv;
      float hn = ov * tanhfast(cn);
      bool  mk = t < lenr[r];
      if (mk) {
        cf [(size_t)b * Hc + col] = cn;
        hf [(size_t)b * Hc + col] = hn;
        hbf[(size_t)b * Hc + col] = f2bf(hn);
      }
      out[(size_t)b * Tc * Hc + (size_t)t * Hc + col] = mk ? hn : 0.0f;
      if (t == Tc - 1) {
        hfin[(size_t)b * Hc + col] = mk ? hn : hf[(size_t)b * Hc + col];
        cfin[(size_t)b * Hc + col] = mk ? cn : cprev;
      }
    }

    // -------- grid-wide step barrier (arrive-release / acquire-spin) ----------
    __threadfence();
    __syncthreads();
    if (tid == 0) {
      __hip_atomic_fetch_add(cnt, 1u, __ATOMIC_RELEASE, __HIP_MEMORY_SCOPE_AGENT);
      unsigned target = (unsigned)(t + 1) * NBLK;
      while (__hip_atomic_load(cnt, __ATOMIC_ACQUIRE, __HIP_MEMORY_SCOPE_AGENT) < target)
        __builtin_amdgcn_s_sleep(1);
    }
    __syncthreads();
  }
}

// ---------------------------------------------------------------------------
extern "C" void kernel_launch(void* const* d_in, const int* in_sizes, int n_in,
                              void* d_out, int out_size, void* d_ws, size_t ws_size,
                              hipStream_t stream) {
  (void)in_sizes; (void)n_in; (void)out_size; (void)ws_size;
  const float*     X    = (const float*)d_in[0];
  const long long* lens = (const long long*)d_in[1];
  const float*     Wih  = (const float*)d_in[2];
  const float*     Whh  = (const float*)d_in[3];
  const float*     bih  = (const float*)d_in[4];
  const float*     bhh  = (const float*)d_in[5];

  char* ws = (char*)d_ws;
  unsigned short* Wc   = (unsigned short*)(ws + OFF_WC);
  float*          bias = (float*)         (ws + OFF_BIAS);
  unsigned short* hbf  = (unsigned short*)(ws + OFF_HBF);
  float*          hf   = (float*)         (ws + OFF_HF);
  float*          cf   = (float*)         (ws + OFF_CF);
  unsigned*       cnt  = (unsigned*)      (ws + OFF_CNT);

  float* out  = (float*)d_out;
  float* hfin = out + (size_t)Bc * Tc * Hc;
  float* cfin = hfin + (size_t)Bc * Hc;

  int prepN = G4 * Kc;                       // 2,097,152 threads cover everything
  lstm_prep<<<(prepN + 255) / 256, 256, 0, stream>>>(Wih, Whh, bih, bhh,
                                                     Wc, bias, hbf, hf, cf, cnt);
  lstm_persist<<<NBLK, 256, 0, stream>>>(X, lens, Wc, bias, hbf, hf, cf, cnt,
                                         out, hfin, cfin);
}
